// LLM_MAIN_66219805769771
// MI455X (gfx1250) — matrix operand order repaired and checked by reading, not compiled, not run on hardware
//
#include <hip/hip_runtime.h>
#include <stdint.h>

// ---------------------------------------------------------------------------
// Types for CDNA5 WMMA (wave32, 16x16x32 bf16 -> f32)
// ---------------------------------------------------------------------------
typedef __attribute__((ext_vector_type(16))) __bf16         bf16x16;
typedef __attribute__((ext_vector_type(8)))  float          f32x8;
typedef __attribute__((ext_vector_type(16))) unsigned short u16x16;
typedef __attribute__((ext_vector_type(8)))  unsigned short u16x8;

// Hardware f32 -> bf16 convert (v_cvt_*_bf16_f32), replaces 4-op integer RNE.
__device__ __forceinline__ unsigned short f2bf(float f) {
  return __builtin_bit_cast(unsigned short, (__bf16)f);
}

// Convert 8 contiguous floats and store as one 16B LDS write (ds_store_b128).
__device__ __forceinline__ void cvt_store8(unsigned short* dst, const float* src) {
  u16x8 v;
#pragma unroll
  for (int j = 0; j < 8; ++j) v[j] = f2bf(src[j]);
  *(u16x8*)dst = v;
}

__device__ __forceinline__ f32x8 zero8() {
  f32x8 z;
#pragma unroll
  for (int i = 0; i < 8; ++i) z[i] = 0.0f;
  return z;
}

// A-matrix fragment (16x32 bf16, MxK). ISA 7.12.2:
// lanes 0-15: M=lane, K in {0..7,16..23}; lanes 16-31: M=lane-16, K in {8..15,24..31}
__device__ __forceinline__ bf16x16 lds_a_frag(const unsigned short* tile, int stride, int lane) {
  const unsigned short* p = tile + (lane & 15) * stride + ((lane & 16) ? 8 : 0);
  u16x8 lo = *(const u16x8*)(p);
  u16x8 hi = *(const u16x8*)(p + 16);
  u16x16 t;
#pragma unroll
  for (int i = 0; i < 8; ++i) { t[i] = lo[i]; t[i + 8] = hi[i]; }
  return __builtin_bit_cast(bf16x16, t);
}

// B-matrix fragment (32x16 bf16, KxN), stored in LDS as rows of N with K contiguous:
// lanes 0-15: N=lane, K in {0..7,16..23}; lanes 16-31: N=lane-16, K in {8..15,24..31}
__device__ __forceinline__ bf16x16 lds_b_frag(const unsigned short* tile, int stride, int lane) {
  const unsigned short* p = tile + (lane & 15) * stride + ((lane & 16) ? 8 : 0);
  u16x8 lo = *(const u16x8*)(p);
  u16x8 hi = *(const u16x8*)(p + 16);
  u16x16 t;
#pragma unroll
  for (int i = 0; i < 8; ++i) { t[i] = lo[i]; t[i + 8] = hi[i]; }
  return __builtin_bit_cast(bf16x16, t);
}

__device__ __forceinline__ f32x8 wmma_bf16(bf16x16 a, bf16x16 b, f32x8 c) {
  return __builtin_amdgcn_wmma_f32_16x16x32_bf16(false, a, false, b, (short)0, c, false, false);
}

// ---------------------------------------------------------------------------
// l2norm over rows: out = x * rsqrt(sum(x^2))   (no epsilon — matches reference)
// ---------------------------------------------------------------------------
__global__ __launch_bounds__(256) void l2norm_rows(const float* __restrict__ x,
                                                   float* __restrict__ out, int ncols) {
  __shared__ float red[8];
  const int row = blockIdx.x, tid = threadIdx.x, lane = tid & 31, wave = tid >> 5;
  const float* xp = x + (size_t)row * ncols;
  float ss = 0.0f;
  for (int i = tid; i < ncols; i += 256) { float v = xp[i]; ss += v * v; }
#pragma unroll
  for (int off = 1; off < 32; off <<= 1) ss += __shfl_xor(ss, off, 32);
  if (lane == 0) red[wave] = ss;
  __syncthreads();
  float tot = 0.0f;
#pragma unroll
  for (int w = 0; w < 8; ++w) tot += red[w];
  const float r = rsqrtf(tot);
  float* op = out + (size_t)row * ncols;
  for (int i = tid; i < ncols; i += 256) op[i] = xp[i] * r;
}

// ---------------------------------------------------------------------------
// C[M,N] = A[M,K] * W[N,K]^T  (+ optional residual), bf16 WMMA, fp32 accumulate
// Block: 128x256 tile, 256 threads = 8 waves (2 along M x 4 along N),
// each wave: 64x64 = 4x4 WMMA tiles (16 WMMA per 8 fragment loads per K-step).
// K staged in steps of 32 through LDS with packed bf16 b128 stores.
// ---------------------------------------------------------------------------
#define LDT 40   // LDS row stride in bf16 elems (multiple of 8 -> 16B aligned loads)

__global__ __launch_bounds__(256) void gemm_l2w(const float* __restrict__ A,
                                                const float* __restrict__ W,
                                                const float* __restrict__ resid,
                                                float* __restrict__ C,
                                                int M, int N, int K) {
  __shared__ unsigned short As[128 * LDT];
  __shared__ unsigned short Bs[256 * LDT];
  const int tid = threadIdx.x, lane = tid & 31, wave = tid >> 5;
  const int bm = blockIdx.y * 128, bn = blockIdx.x * 256;
  const int wm = (wave >> 2) * 64, wn = (wave & 3) * 64;
  const int arow = tid >> 1, aseg = (tid & 1) * 16;      // A: half row per thread
  const float* ap = A + (size_t)(bm + arow) * K + aseg;
  const float* wp = W + (size_t)(bn + tid) * K;          // B: full 32-wide row per thread

  f32x8 acc[4][4];
#pragma unroll
  for (int mt = 0; mt < 4; ++mt)
#pragma unroll
    for (int nt = 0; nt < 4; ++nt) acc[mt][nt] = zero8();

  for (int k0 = 0; k0 < K; k0 += 32) {
#pragma unroll
    for (int i = 0; i < 16; i += 8)
      cvt_store8(&As[arow * LDT + aseg + i], ap + k0 + i);
#pragma unroll
    for (int i = 0; i < 32; i += 8)
      cvt_store8(&Bs[tid * LDT + i], wp + k0 + i);
    if (k0 + 32 < K) {           // prefetch next K tile (global_prefetch_b8)
      __builtin_prefetch(ap + k0 + 32, 0, 1);
      __builtin_prefetch(wp + k0 + 32, 0, 1);
    }
    __syncthreads();
    bf16x16 bfr[4];
#pragma unroll
    for (int nt = 0; nt < 4; ++nt)
      bfr[nt] = lds_b_frag(Bs + (wn + nt * 16) * LDT, LDT, lane);
#pragma unroll
    for (int mt = 0; mt < 4; ++mt) {
      bf16x16 afr = lds_a_frag(As + (wm + mt * 16) * LDT, LDT, lane);
#pragma unroll
      for (int nt = 0; nt < 4; ++nt) acc[mt][nt] = wmma_bf16(afr, bfr[nt], acc[mt][nt]);
    }
    __syncthreads();
  }

  const int hf = (lane >> 4) * 8, lc = lane & 15;   // C layout: VGPR v -> M = v (+8 hi half)
#pragma unroll
  for (int mt = 0; mt < 4; ++mt)
#pragma unroll
    for (int nt = 0; nt < 4; ++nt)
#pragma unroll
      for (int v = 0; v < 8; ++v) {
        const int row = bm + wm + mt * 16 + v + hf;
        const int col = bn + wn + nt * 16 + lc;
        float r = acc[mt][nt][v];
        if (resid) r += resid[(size_t)row * N + col];
        C[(size_t)row * N + col] = r;
      }
}

// ---------------------------------------------------------------------------
// Per-head l2norm * qk_norm_w, then RoPE (sin sign pre-baked), split Q/K, copy V
// ---------------------------------------------------------------------------
__global__ __launch_bounds__(256) void qk_rope_kernel(const float* __restrict__ qkv,
                                                      const float* __restrict__ qknw,
                                                      const float* __restrict__ cosp,
                                                      const float* __restrict__ sinp,
                                                      float* __restrict__ Q,
                                                      float* __restrict__ Kg,
                                                      float* __restrict__ Vg) {
  const int s = blockIdx.x, tid = threadIdx.x, lane = tid & 31, wave = tid >> 5;
  const float* cs = cosp + (size_t)s * 128;
  const float* sn = sinp + (size_t)s * 128;
#pragma unroll
  for (int it = 0; it < 3; ++it) {
    const int head = wave + it * 8;                      // 0..23
    const float* xp = qkv + (size_t)s * 4096 + head * 128;
    const float* wp = qknw + head * 128;
    const int d = lane;                                  // lane handles d, d+32, d+64, d+96
    float x0 = xp[d], x1 = xp[d + 32], x2 = xp[d + 64], x3 = xp[d + 96];
    float ss = x0 * x0 + x1 * x1 + x2 * x2 + x3 * x3;
#pragma unroll
    for (int off = 1; off < 32; off <<= 1) ss += __shfl_xor(ss, off, 32);
    const float r = rsqrtf(ss);
    const float n0 = x0 * r * wp[d];
    const float n1 = x1 * r * wp[d + 32];
    const float n2 = x2 * r * wp[d + 64];
    const float n3 = x3 * r * wp[d + 96];
    const float o0 = n0 * cs[d]      + n2 * sn[d];       // rot = concat(x2,x1)
    const float o1 = n1 * cs[d + 32] + n3 * sn[d + 32];
    const float o2 = n2 * cs[d + 64] + n0 * sn[d + 64];
    const float o3 = n3 * cs[d + 96] + n1 * sn[d + 96];
    float* outp = (head < 16) ? (Q  + ((size_t)s * 16 + head) * 128)
                              : (Kg + ((size_t)s * 8 + (head - 16)) * 128);
    outp[d] = o0; outp[d + 32] = o1; outp[d + 64] = o2; outp[d + 96] = o3;
  }
  for (int i = tid; i < 1024; i += 256)
    Vg[(size_t)s * 1024 + i] = qkv[(size_t)s * 4096 + 3072 + i];
}

// ---------------------------------------------------------------------------
// Flash attention, one block per (head, 128-query tile). 8 waves x 16 q-rows.
// Additive causal term -128*mask (matches reference, NOT -inf), online softmax.
// Both GEMMs (Q K^T and P V) are WMMA bf16.
// ---------------------------------------------------------------------------
__global__ __launch_bounds__(256) void attn_kernel(const float* __restrict__ Q,
                                                   const float* __restrict__ Kg,
                                                   const float* __restrict__ Vg,
                                                   const float* __restrict__ maskp,
                                                   float* __restrict__ AO) {
  __shared__ unsigned short Ks[64 * 136];   // keys x d, bf16
  __shared__ unsigned short Vt[128 * 72];   // d x keys (transposed), bf16
  __shared__ unsigned short Ps[8][16 * 72]; // per-wave P tile, bf16
  const int tid = threadIdx.x, lane = tid & 31, wave = tid >> 5;
  const int h = blockIdx.x, kvh = h >> 1;   // GROUPS = 2
  const int qbase = blockIdx.y * 128 + wave * 16;
  const float negadd = -128.0f * maskp[0];
  const int lc = lane & 15, hf = (lane >> 4) * 8;

  // Q fragments (A layout), 4 K-steps of 32 over HD=128, fp32 -> bf16
  bf16x16 aq[4];
#pragma unroll
  for (int ks = 0; ks < 4; ++ks) {
    const int qrow = qbase + lc;
    const float* qp = Q + ((size_t)qrow * 16 + h) * 128 + ks * 32 + ((lane & 16) ? 8 : 0);
    u16x16 t;
#pragma unroll
    for (int i = 0; i < 8; ++i) { t[i] = f2bf(qp[i]); t[i + 8] = f2bf(qp[16 + i]); }
    aq[ks] = __builtin_bit_cast(bf16x16, t);
  }

  float m[8], l[8];
  f32x8 o[8];
#pragma unroll
  for (int v = 0; v < 8; ++v) { m[v] = -1e30f; l[v] = 0.0f; }
#pragma unroll
  for (int dt = 0; dt < 8; ++dt) o[dt] = zero8();

  for (int kt = 0; kt < 16; ++kt) {          // 16 key tiles of 64
    {
      const int r = tid >> 2, seg = (tid & 3) * 32;
      const float* kp = Kg + ((size_t)(kt * 64 + r) * 8 + kvh) * 128 + seg;
      const float* vp = Vg + ((size_t)(kt * 64 + r) * 8 + kvh) * 128 + seg;
#pragma unroll
      for (int i = 0; i < 32; i += 8)        // packed b128 stores for K tile
        cvt_store8(&Ks[r * 136 + seg + i], kp + i);
#pragma unroll 8
      for (int i = 0; i < 32; ++i)           // transpose-on-store for V tile
        Vt[(seg + i) * 72 + r] = f2bf(vp[i]);
    }
    __syncthreads();

    // S = Q K^T : 4 x 16-key tiles, K-dim = HD in 4 steps of 32
    f32x8 sc[4];
#pragma unroll
    for (int nt = 0; nt < 4; ++nt) sc[nt] = zero8();
#pragma unroll
    for (int ks = 0; ks < 4; ++ks)
#pragma unroll
      for (int nt = 0; nt < 4; ++nt) {
        bf16x16 b = lds_b_frag(Ks + (nt * 16) * 136 + ks * 32, 136, lane);
        sc[nt] = wmma_bf16(aq[ks], b, sc[nt]);
      }

    // causal additive mask + online softmax (per-row stats, 16-lane-half shuffles)
    float scl[8];
#pragma unroll
    for (int v = 0; v < 8; ++v) {
      const int row = qbase + v + hf;
      float mx = -1e30f;
#pragma unroll
      for (int nt = 0; nt < 4; ++nt) {
        const int key = kt * 64 + nt * 16 + lc;
        float s = sc[nt][v];
        if (key > row) s += negadd;
        sc[nt][v] = s;
        mx = fmaxf(mx, s);
      }
#pragma unroll
      for (int off = 1; off < 16; off <<= 1) mx = fmaxf(mx, __shfl_xor(mx, off, 32));
      const float mnew = fmaxf(m[v], mx);
      const float esc  = __expf(m[v] - mnew);
      float rs = 0.0f;
#pragma unroll
      for (int nt = 0; nt < 4; ++nt) {
        float p = __expf(sc[nt][v] - mnew);
        sc[nt][v] = p;
        rs += p;
      }
#pragma unroll
      for (int off = 1; off < 16; off <<= 1) rs += __shfl_xor(rs, off, 32);
      l[v] = l[v] * esc + rs;
      m[v] = mnew;
      scl[v] = esc;
    }
#pragma unroll
    for (int dt = 0; dt < 8; ++dt)
#pragma unroll
      for (int v = 0; v < 8; ++v) o[dt][v] *= scl[v];

    // P (C layout) -> LDS -> A layout for the P*V WMMA
    unsigned short* pw = Ps[wave];
#pragma unroll
    for (int nt = 0; nt < 4; ++nt)
#pragma unroll
      for (int v = 0; v < 8; ++v)
        pw[(v + hf) * 72 + nt * 16 + lc] = f2bf(sc[nt][v]);
    __asm__ __volatile__("" ::: "memory");   // keep DS write->read order (same-wave, in-order)

    // O += P * V : K-dim = 64 keys in 2 steps of 32, N = HD in 8 tiles
#pragma unroll
    for (int ks2 = 0; ks2 < 2; ++ks2) {
      bf16x16 pf = lds_a_frag(pw + ks2 * 32, 72, lane);
#pragma unroll
      for (int dt = 0; dt < 8; ++dt) {
        bf16x16 b = lds_b_frag(Vt + (dt * 16) * 72 + ks2 * 32, 72, lane);
        o[dt] = wmma_bf16(pf, b, o[dt]);
      }
    }
    __syncthreads();
  }

#pragma unroll
  for (int dt = 0; dt < 8; ++dt)
#pragma unroll
    for (int v = 0; v < 8; ++v) {
      const int row = qbase + v + hf;
      const int d = dt * 16 + lc;
      AO[(size_t)row * 2048 + h * 128 + d] = o[dt][v] / l[v];
    }
}

// ---------------------------------------------------------------------------
// act = silu(gate) * up
// ---------------------------------------------------------------------------
__global__ __launch_bounds__(256) void silu_mul(const float* __restrict__ gu,
                                                float* __restrict__ act) {
  const int idx = blockIdx.x * 256 + threadIdx.x;   // grid covers S*DFF exactly
  const int s = idx / 6144, f = idx - s * 6144;
  const float g = gu[(size_t)s * 12288 + f];
  const float u = gu[(size_t)s * 12288 + 6144 + f];
  act[idx] = g / (1.0f + __expf(-g)) * u;
}

// ---------------------------------------------------------------------------
// logits[row] = dot(ln, lm_head_w[row])  — bandwidth-bound GEMV, wave per row
// ---------------------------------------------------------------------------
__global__ __launch_bounds__(256) void lmhead(const float* __restrict__ ln,
                                              const float* __restrict__ W,
                                              float* __restrict__ out) {
  const int tid = threadIdx.x, lane = tid & 31, wave = tid >> 5;
  const int row = blockIdx.x * 8 + wave;
  const float* wp = W + (size_t)row * 2048;
  float s = 0.0f;
  for (int i = lane; i < 2048; i += 32) s += ln[i] * wp[i];
#pragma unroll
  for (int off = 1; off < 32; off <<= 1) s += __shfl_xor(s, off, 32);
  if (lane == 0) out[row] = s;
}

// ---------------------------------------------------------------------------
extern "C" void kernel_launch(void* const* d_in, const int* in_sizes, int n_in,
                              void* d_out, int out_size, void* d_ws, size_t ws_size,
                              hipStream_t stream) {
  (void)in_sizes; (void)n_in; (void)out_size; (void)ws_size;
  const float* hs        = (const float*)d_in[0];
  const float* qkv_w     = (const float*)d_in[1];
  const float* qk_norm_w = (const float*)d_in[2];
  const float* o_w       = (const float*)d_in[3];
  const float* gate_up_w = (const float*)d_in[4];
  const float* down_w    = (const float*)d_in[5];
  const float* lm_head_w = (const float*)d_in[6];
  const float* cosp      = (const float*)d_in[7];
  const float* sinp      = (const float*)d_in[8];
  const float* maskp     = (const float*)d_in[9];

  float* p   = (float*)d_ws;
  float* X   = p; p += (size_t)1024 * 2048;
  float* HN  = p; p += (size_t)1024 * 2048;
  float* QKV = p; p += (size_t)1024 * 4096;
  float* Qb  = p; p += (size_t)1024 * 2048;
  float* Kb  = p; p += (size_t)1024 * 1024;
  float* Vb  = p; p += (size_t)1024 * 1024;
  float* AO  = p; p += (size_t)1024 * 2048;
  float* GU  = p; p += (size_t)1024 * 12288;
  float* ACT = p; p += (size_t)1024 * 6144;
  float* LN  = p; p += 2048;

  hipMemcpyAsync(X, hs, (size_t)1024 * 2048 * sizeof(float),
                 hipMemcpyDeviceToDevice, stream);

  for (int l = 0; l < 2; ++l) {
    l2norm_rows<<<1024, 256, 0, stream>>>(X, HN, 2048);
    gemm_l2w<<<dim3(16, 8), 256, 0, stream>>>(HN, qkv_w + (size_t)l * 4096 * 2048,
                                              nullptr, QKV, 1024, 4096, 2048);
    qk_rope_kernel<<<1024, 256, 0, stream>>>(QKV, qk_norm_w + (size_t)l * 24 * 128,
                                             cosp, sinp, Qb, Kb, Vb);
    attn_kernel<<<dim3(16, 8), 256, 0, stream>>>(Qb, Kb, Vb, maskp, AO);
    gemm_l2w<<<dim3(8, 8), 256, 0, stream>>>(AO, o_w + (size_t)l * 2048 * 2048,
                                             X, X, 1024, 2048, 2048);
    l2norm_rows<<<1024, 256, 0, stream>>>(X, HN, 2048);
    gemm_l2w<<<dim3(48, 8), 256, 0, stream>>>(HN, gate_up_w + (size_t)l * 12288 * 2048,
                                              nullptr, GU, 1024, 12288, 2048);
    silu_mul<<<(1024 * 6144) / 256, 256, 0, stream>>>(GU, ACT);
    gemm_l2w<<<dim3(8, 8), 256, 0, stream>>>(ACT, down_w + (size_t)l * 2048 * 6144,
                                             X, X, 1024, 2048, 6144);
  }
  l2norm_rows<<<1, 256, 0, stream>>>(X + (size_t)1023 * 2048, LN, 2048);
  lmhead<<<4000, 256, 0, stream>>>(LN, lm_head_w, (float*)d_out);
}
